// HRNN_simple_65042984730676
// MI455X (gfx1250) — matrix operand order repaired and checked by reading, not compiled, not run on hardware
//
#include <hip/hip_runtime.h>

#define EMB     64
#define SEG     10
#define CLICK   20
#define LONG_T  20
#define SHORT_T 10
#define WORDN   50000
#define ITEMN   100000
#define BROWS   16

typedef __attribute__((ext_vector_type(16))) _Float16 v16h;
typedef __attribute__((ext_vector_type(8)))  float    v8f;

// ---------------- fast activations (branch-free) ----------------

__device__ __forceinline__ float fast_tanh(float x) {
#if __has_builtin(__builtin_amdgcn_tanhf)
  return __builtin_amdgcn_tanhf(x);       // native v_tanh_f32 on gfx1250
#else
  return 1.f - 2.f / (1.f + __expf(2.f * x));
#endif
}

__device__ __forceinline__ float fast_sigmoid(float x) {
  return 1.f / (1.f + __expf(-x));        // native v_exp_f32, branch-free
}

// ---------------- WMMA helpers (gfx1250, wave32) ----------------

__device__ __forceinline__ v8f wmma_f16(v16h a, v16h b, v8f c) {
  // D = A(16x32 f16) * B(32x16 f16) + C(16x16 f32)
  return __builtin_amdgcn_wmma_f32_16x16x32_f16(
      /*neg_a=*/false, a, /*neg_b=*/false, b,
      /*c_mod=*/(short)0, c, /*reuse_a=*/false, /*reuse_b=*/false);
}

// A-matrix fragment from an LDS f16 buffer laid out row-major [M][ldm].
// Per ISA 16-bit A 16x32 layout: lane L row = L&15;
//   elements 0..7  -> K = k0 + (L>>4)*8 + e
//   elements 8..15 -> K = k0 + (L>>4)*8 + 16 + (e-8)
__device__ __forceinline__ v16h ldsA(const _Float16* base, int ldm, int k0, int lane) {
  const _Float16* p = base + (lane & 15) * ldm + k0 + ((lane >> 4) << 3);
  v16h a;
#pragma unroll
  for (int e = 0; e < 8; ++e) { a[e] = p[e]; a[e + 8] = p[16 + e]; }
  return a;
}

// B-matrix fragment B[k][n] = W[n][k] built from a row-major f32 weight
// W[N][ldk] in global memory (converted to f16 in-register).
// Lane L col n = n0 + (L&15); element e<8 -> k = k0 + (L>>4)*8 + e, e>=8 -> k = k0 + (L>>4)*8 + 16 + (e-8).
__device__ __forceinline__ v16h gblB(const float* W, int ldk, int n0, int k0, int lane) {
  const float* p = W + (size_t)(n0 + (lane & 15)) * ldk + k0 + ((lane >> 4) << 3);
  v16h b;
#pragma unroll
  for (int e = 0; e < 8; ++e) { b[e] = (_Float16)p[e]; b[e + 8] = (_Float16)p[16 + e]; }
  return b;
}

// ---------------- Kernel 0: collapse attention MLP ----------------
// e = tanh((qs@W1^T+b1)@W2^T+b2) == tanh(qs . w_eff + b_eff)  (inner is linear)
__global__ void k_att_collapse(const float* __restrict__ W1, const float* __restrict__ b1,
                               const float* __restrict__ W2, const float* __restrict__ b2,
                               float* __restrict__ w_eff, float* __restrict__ b_eff) {
  int j = threadIdx.x;  // 128 threads
  float acc = 0.f;
  for (int s = 0; s < 512; ++s) acc += W2[s] * W1[s * 128 + j];
  w_eff[j] = acc;
  if (j == 0) {
    float bb = 0.f;
    for (int s = 0; s < 512; ++s) bb += W2[s] * b1[s];
    b_eff[0] = bb + b2[0];
  }
}

// ---------------- Kernel 1: pooling + qproj (WMMA) ----------------
__global__ void k_pool_qproj(
    const int* __restrict__ query_ids, const int* __restrict__ query_seg,
    const int* __restrict__ query_seg_lens, const int* __restrict__ query_click,
    const int* __restrict__ query_click_lens, const int* __restrict__ query_long_his,
    const int* __restrict__ query_short_his,
    const float* __restrict__ word_emb, const float* __restrict__ item_emb,
    const float* __restrict__ Wq, const float* __restrict__ bq,
    _Float16* __restrict__ q_short, _Float16* __restrict__ i_short,
    _Float16* __restrict__ q_long, _Float16* __restrict__ i_long,
    float* __restrict__ q_self) {
  extern __shared__ char smem[];
  _Float16* qpool = (_Float16*)smem;  // [31][16][64] f16 = 63488 B

  const int tid = threadIdx.x, lane = tid & 31;
  const int wave = __builtin_amdgcn_readfirstlane(tid >> 5);  // force SGPR
  const int b0 = blockIdx.x * BROWS;

  // ---- q-type pooling (masked mean over up to SEG word embeddings) ----
  for (int idx = tid; idx < 31 * BROWS * EMB; idx += blockDim.x) {
    int e = idx & 63, row = (idx >> 6) & 15, slot = idx >> 10;
    int qb = query_ids[b0 + row];
    int qid;
    if (slot < SHORT_T)      qid = query_short_his[qb * SHORT_T + slot];
    else if (slot < 30)      qid = query_long_his[qb * LONG_T + (slot - SHORT_T)];
    else                     qid = qb;
    int len = query_seg_lens[qid];
    float acc = 0.f;
    for (int t = 0; t < len; ++t) {
      int w = query_seg[qid * SEG + t];
      if (w < WORDN) acc += word_emb[(size_t)w * EMB + e];  // id==WORDN -> zero row
    }
    qpool[(slot * BROWS + row) * EMB + e] = (_Float16)(acc / (float)len);
  }

  // ---- i-type pooling (masked mean over up to CLICK item embeddings) ----
  for (int idx = tid; idx < 30 * BROWS * EMB; idx += blockDim.x) {
    int e = idx & 63, row = (idx >> 6) & 15, slot = idx >> 10;
    int b = b0 + row;
    int qb = query_ids[b];
    int qid = (slot < SHORT_T) ? query_short_his[qb * SHORT_T + slot]
                               : query_long_his[qb * LONG_T + (slot - SHORT_T)];
    int len = query_click_lens[qid];
    float acc = 0.f;
    for (int t = 0; t < len; ++t) {
      int it = query_click[qid * CLICK + t];
      if (it < ITEMN) acc += item_emb[(size_t)it * EMB + e];
    }
    _Float16 v = (_Float16)(acc / (float)len);
    if (slot < SHORT_T) i_short[((size_t)b * SHORT_T + slot) * EMB + e] = v;
    else                i_long[((size_t)b * LONG_T + (slot - SHORT_T)) * EMB + e] = v;
  }

  // preload all 8 Wq B-fragments (statically indexed below -> stay in VGPRs)
  v16h bWq[8];
#pragma unroll
  for (int f = 0; f < 8; ++f)
    bWq[f] = gblB(Wq, EMB, (f >> 1) * 16, (f & 1) * 32, lane);

  // per-lane biases for the 4 n-tiles, hoisted out of the slot loop
  float bias[4];
#pragma unroll
  for (int nt = 0; nt < 4; ++nt) bias[nt] = bq[nt * 16 + (lane & 15)];

  __syncthreads();

  // ---- qproj: y = tanh(x @ Wq^T + bq); one slot (16x64 tile) per wave-job ----
  const int nwaves = blockDim.x >> 5;
  for (int slot = wave; slot < 31; slot += nwaves) {   // slot is scalar
    const _Float16* ap = qpool + slot * BROWS * EMB;
    v16h a0 = ldsA(ap, EMB, 0, lane);
    v16h a1 = ldsA(ap, EMB, 32, lane);
    v8f c[4];
#pragma unroll
    for (int nt = 0; nt < 4; ++nt) {
      v8f z = {};
      z = wmma_f16(a0, bWq[nt * 2 + 0], z);
      c[nt] = wmma_f16(a1, bWq[nt * 2 + 1], z);
    }
    // scalar-branch store paths (slot uniform)
    if (slot < SHORT_T) {
#pragma unroll
      for (int nt = 0; nt < 4; ++nt) {
        int n = nt * 16 + (lane & 15);
#pragma unroll
        for (int r = 0; r < 8; ++r) {
          int m = r + ((lane >> 4) << 3);
          q_short[((size_t)(b0 + m) * SHORT_T + slot) * EMB + n] =
              (_Float16)fast_tanh(c[nt][r] + bias[nt]);
        }
      }
    } else if (slot < 30) {
      int sl = slot - SHORT_T;
#pragma unroll
      for (int nt = 0; nt < 4; ++nt) {
        int n = nt * 16 + (lane & 15);
#pragma unroll
        for (int r = 0; r < 8; ++r) {
          int m = r + ((lane >> 4) << 3);
          q_long[((size_t)(b0 + m) * LONG_T + sl) * EMB + n] =
              (_Float16)fast_tanh(c[nt][r] + bias[nt]);
        }
      }
    } else {
#pragma unroll
      for (int nt = 0; nt < 4; ++nt) {
        int n = nt * 16 + (lane & 15);
#pragma unroll
        for (int r = 0; r < 8; ++r) {
          int m = r + ((lane >> 4) << 3);
          q_self[(size_t)(b0 + m) * EMB + n] = fast_tanh(c[nt][r] + bias[nt]);
        }
      }
    }
  }
}

// ---------------- Kernel 2: short GRU + projection ----------------
// 12 waves; wave w owns gate columns [16w,16w+16) with Wih^T/Whh^T fragments
// resident in VGPRs across all 10 timesteps.
__global__ void k_gru_short(
    const int* __restrict__ query_ids, const int* __restrict__ short_lens,
    const _Float16* __restrict__ q_short, const _Float16* __restrict__ i_short,
    const float* __restrict__ Wih, const float* __restrict__ Whh,
    const float* __restrict__ bih, const float* __restrict__ bhh,
    const float* __restrict__ short_proj, float* __restrict__ out) {
  extern __shared__ char smem[];
  _Float16* x16  = (_Float16*)smem;              // [16][128]
  _Float16* h16  = x16 + BROWS * 128;            // [16][64]
  float* gi      = (float*)(h16 + BROWS * 64);   // [16][192]
  float* gh      = gi + BROWS * 192;             // [16][192]
  float* hbuf    = gh + BROWS * 192;             // [16][64]
  float* interest= hbuf + BROWS * 64;            // [16][64]
  int*  lastt    = (int*)(interest + BROWS * 64);// [16]
  _Float16* i16b = (_Float16*)(lastt + 16);      // [16][64]

  const int tid = threadIdx.x, lane = tid & 31;
  const int wave = __builtin_amdgcn_readfirstlane(tid >> 5);  // force SGPR
  const int b0 = blockIdx.x * BROWS;

  for (int i = tid; i < BROWS * 64; i += blockDim.x) { hbuf[i] = 0.f; h16[i] = (_Float16)0.f; }
  if (tid < BROWS) lastt[tid] = short_lens[query_ids[b0 + tid]] - 1;

  v16h bWih[4], bWhh[2];
#pragma unroll
  for (int kt = 0; kt < 4; ++kt) bWih[kt] = gblB(Wih, 128, wave * 16, kt * 32, lane);
#pragma unroll
  for (int kt = 0; kt < 2; ++kt) bWhh[kt] = gblB(Whh, 64, wave * 16, kt * 32, lane);
  __syncthreads();

  for (int t = 0; t < SHORT_T; ++t) {
    // x_t = [q | i] as 16-byte chunks: 16 rows x 16 chunks
    if (tid < BROWS * 16) {
      int row = tid >> 4, c = tid & 15;
      size_t b = (size_t)(b0 + row);
      const uint4* src = (c < 8)
          ? (const uint4*)(q_short + (b * SHORT_T + t) * EMB) + c
          : (const uint4*)(i_short + (b * SHORT_T + t) * EMB) + (c - 8);
      *((uint4*)(x16 + row * 128) + c) = *src;
    }
    __syncthreads();

    v8f cgi = {}, cgh = {};
#pragma unroll
    for (int kt = 0; kt < 4; ++kt) cgi = wmma_f16(ldsA(x16, 128, kt * 32, lane), bWih[kt], cgi);
#pragma unroll
    for (int kt = 0; kt < 2; ++kt) cgh = wmma_f16(ldsA(h16, 64, kt * 32, lane), bWhh[kt], cgh);
    {
      int n = wave * 16 + (lane & 15);
#pragma unroll
      for (int r = 0; r < 8; ++r) {
        int m = r + ((lane >> 4) << 3);
        gi[m * 192 + n] = cgi[r];
        gh[m * 192 + n] = cgh[r];
      }
    }
    __syncthreads();

    for (int idx = tid; idx < BROWS * 64; idx += blockDim.x) {
      int m = idx >> 6, j = idx & 63;
      float ir = gi[m * 192 + j]        + bih[j];
      float iz = gi[m * 192 + 64 + j]   + bih[64 + j];
      float in_= gi[m * 192 + 128 + j]  + bih[128 + j];
      float hr = gh[m * 192 + j]        + bhh[j];
      float hz = gh[m * 192 + 64 + j]   + bhh[64 + j];
      float hn = gh[m * 192 + 128 + j]  + bhh[128 + j];
      float rg = fast_sigmoid(ir + hr);
      float zg = fast_sigmoid(iz + hz);
      float ng = fast_tanh(in_ + rg * hn);
      float hnew = (1.f - zg) * ng + zg * hbuf[idx];
      hbuf[idx] = hnew;
      h16[idx] = (_Float16)hnew;
      if (t == lastt[m]) interest[idx] = hnew;
    }
    __syncthreads();
  }

  for (int i = tid; i < BROWS * 64; i += blockDim.x) i16b[i] = (_Float16)interest[i];
  __syncthreads();

  if (wave < 4) {  // scalar branch; EXEC all-ones inside
    int nt = wave;
    v8f c = {};
#pragma unroll
    for (int kt = 0; kt < 2; ++kt)
      c = wmma_f16(ldsA(i16b, 64, kt * 32, lane),
                   gblB(short_proj, 64, nt * 16, kt * 32, lane), c);
    int n = nt * 16 + (lane & 15);
#pragma unroll
    for (int r = 0; r < 8; ++r) {
      int m = r + ((lane >> 4) << 3);
      out[(size_t)(b0 + m) * 128 + n] = c[r];
    }
  }
}

// ---------------- Kernel 3: long GRU + attention + projection ----------------
__global__ void k_gru_long(
    const int* __restrict__ query_ids, const int* __restrict__ long_lens,
    const _Float16* __restrict__ q_long, const _Float16* __restrict__ i_long,
    const float* __restrict__ Wih, const float* __restrict__ Whh,
    const float* __restrict__ bih, const float* __restrict__ bhh,
    const float* __restrict__ q_self, const float* __restrict__ w_eff,
    const float* __restrict__ b_eff, const float* __restrict__ long_proj,
    float* __restrict__ out) {
  extern __shared__ char smem[];
  _Float16* x16   = (_Float16*)smem;               // [16][128]
  _Float16* h16   = x16 + BROWS * 128;             // [16][64]
  float* gi       = (float*)(h16 + BROWS * 64);    // [16][192]
  float* gh       = gi + BROWS * 192;              // [16][192]
  float* hbuf     = gh + BROWS * 192;              // [16][64]
  float* e_a      = hbuf + BROWS * 64;             // [16][LONG_T]
  _Float16* i16b  = (_Float16*)(e_a + BROWS * LONG_T);  // [16][64]
  _Float16* hist16= i16b + BROWS * 64;             // [20][16][64] f16 = 40960 B

  const int tid = threadIdx.x, lane = tid & 31;
  const int wave = __builtin_amdgcn_readfirstlane(tid >> 5);  // force SGPR
  const int b0 = blockIdx.x * BROWS;

  for (int i = tid; i < BROWS * 64; i += blockDim.x) { hbuf[i] = 0.f; h16[i] = (_Float16)0.f; }

  // Per-row constants (thread m owns row m for the scalar attention work).
  float c0r = 0.f, be = 0.f;
  int mylen = 0;
  if (tid < BROWS) {
    be = b_eff[0];
    mylen = long_lens[query_ids[b0 + tid]];
    const float* qs = q_self + (size_t)(b0 + tid) * EMB;
    for (int j = 0; j < 64; ++j) c0r += qs[j] * w_eff[j];
  }

  v16h bWih[4], bWhh[2];
#pragma unroll
  for (int kt = 0; kt < 4; ++kt) bWih[kt] = gblB(Wih, 128, wave * 16, kt * 32, lane);
#pragma unroll
  for (int kt = 0; kt < 2; ++kt) bWhh[kt] = gblB(Whh, 64, wave * 16, kt * 32, lane);
  __syncthreads();

  for (int t = 0; t < LONG_T; ++t) {
    if (tid < BROWS * 16) {
      int row = tid >> 4, c = tid & 15;
      size_t b = (size_t)(b0 + row);
      const uint4* src = (c < 8)
          ? (const uint4*)(q_long + (b * LONG_T + t) * EMB) + c
          : (const uint4*)(i_long + (b * LONG_T + t) * EMB) + (c - 8);
      *((uint4*)(x16 + row * 128) + c) = *src;
    }
    __syncthreads();

    v8f cgi = {}, cgh = {};
#pragma unroll
    for (int kt = 0; kt < 4; ++kt) cgi = wmma_f16(ldsA(x16, 128, kt * 32, lane), bWih[kt], cgi);
#pragma unroll
    for (int kt = 0; kt < 2; ++kt) cgh = wmma_f16(ldsA(h16, 64, kt * 32, lane), bWhh[kt], cgh);
    {
      int n = wave * 16 + (lane & 15);
#pragma unroll
      for (int r = 0; r < 8; ++r) {
        int m = r + ((lane >> 4) << 3);
        gi[m * 192 + n] = cgi[r];
        gh[m * 192 + n] = cgh[r];
      }
    }
    __syncthreads();

    for (int idx = tid; idx < BROWS * 64; idx += blockDim.x) {
      int m = idx >> 6, j = idx & 63;
      float ir = gi[m * 192 + j]        + bih[j];
      float iz = gi[m * 192 + 64 + j]   + bih[64 + j];
      float in_= gi[m * 192 + 128 + j]  + bih[128 + j];
      float hr = gh[m * 192 + j]        + bhh[j];
      float hz = gh[m * 192 + 64 + j]   + bhh[64 + j];
      float hn = gh[m * 192 + 128 + j]  + bhh[128 + j];
      float rg = fast_sigmoid(ir + hr);
      float zg = fast_sigmoid(iz + hz);
      float ng = fast_tanh(in_ + rg * hn);
      float hnew = (1.f - zg) * ng + zg * hbuf[idx];
      hbuf[idx] = hnew;
      h16[idx] = (_Float16)hnew;
      hist16[(t * BROWS + m) * 64 + j] = (_Float16)hnew;
    }
    __syncthreads();

    // e_t[m] = tanh(c0[m] + h_t[m] . w_eff[64:128] + b_eff)
    if (tid < BROWS) {
      float acc = c0r;
      for (int j = 0; j < 64; ++j) acc += hbuf[tid * 64 + j] * w_eff[64 + j];
      e_a[tid * LONG_T + t] = fast_tanh(acc + be);
    }
    // next x16 load touches different LDS; a barrier follows before gi/gh reuse
  }
  __syncthreads();

  // masked softmax over t (pad positions get weight 0, matching softmax(-1e9))
  if (tid < BROWS) {
    float mx = -1e30f;
    for (int t = 0; t < mylen; ++t) mx = fmaxf(mx, e_a[tid * LONG_T + t]);
    float s = 0.f;
    for (int t = 0; t < mylen; ++t) {
      float ex = __expf(e_a[tid * LONG_T + t] - mx);
      e_a[tid * LONG_T + t] = ex;
      s += ex;
    }
    float inv = 1.f / s;
    for (int t = 0; t < LONG_T; ++t)
      e_a[tid * LONG_T + t] = (t < mylen) ? e_a[tid * LONG_T + t] * inv : 0.f;
  }
  __syncthreads();

  // long_interest = sum_t a[t] * h_t   (f16 history, f32 accumulate)
  for (int idx = tid; idx < BROWS * 64; idx += blockDim.x) {
    int m = idx >> 6, j = idx & 63;
    float acc = 0.f;
    for (int t = 0; t < LONG_T; ++t)
      acc += e_a[m * LONG_T + t] * (float)hist16[(t * BROWS + m) * 64 + j];
    i16b[idx] = (_Float16)acc;
  }
  __syncthreads();

  if (wave < 4) {  // scalar branch
    int nt = wave;
    v8f c = {};
#pragma unroll
    for (int kt = 0; kt < 2; ++kt)
      c = wmma_f16(ldsA(i16b, 64, kt * 32, lane),
                   gblB(long_proj, 64, nt * 16, kt * 32, lane), c);
    int n = nt * 16 + (lane & 15);
#pragma unroll
    for (int r = 0; r < 8; ++r) {
      int m = r + ((lane >> 4) << 3);
      out[(size_t)(b0 + m) * 128 + 64 + n] = c[r];
    }
  }
}

// ---------------- host launch ----------------
extern "C" void kernel_launch(void* const* d_in, const int* in_sizes, int n_in,
                              void* d_out, int out_size, void* d_ws, size_t ws_size,
                              hipStream_t stream) {
  const int* query_ids        = (const int*)d_in[0];
  const int* query_seg        = (const int*)d_in[1];
  const int* query_seg_lens   = (const int*)d_in[2];
  const int* query_click      = (const int*)d_in[3];
  const int* query_click_lens = (const int*)d_in[4];
  const int* query_long_his   = (const int*)d_in[5];
  const int* query_short_his  = (const int*)d_in[6];
  const int* long_lens        = (const int*)d_in[7];
  const int* short_lens       = (const int*)d_in[8];
  const float* word_emb   = (const float*)d_in[9];
  const float* item_emb   = (const float*)d_in[10];
  const float* Wq         = (const float*)d_in[11];
  const float* bq         = (const float*)d_in[12];
  const float* att_W1     = (const float*)d_in[13];
  const float* att_b1     = (const float*)d_in[14];
  const float* att_W2     = (const float*)d_in[15];
  const float* att_b2     = (const float*)d_in[16];
  const float* s_Wih      = (const float*)d_in[17];
  const float* s_Whh      = (const float*)d_in[18];
  const float* s_bih      = (const float*)d_in[19];
  const float* s_bhh      = (const float*)d_in[20];
  const float* l_Wih      = (const float*)d_in[21];
  const float* l_Whh      = (const float*)d_in[22];
  const float* l_bih      = (const float*)d_in[23];
  const float* l_bhh      = (const float*)d_in[24];
  const float* short_proj = (const float*)d_in[25];
  const float* long_proj  = (const float*)d_in[26];

  const int B = in_sizes[0];       // 4096
  const int nblk = B / BROWS;      // 256

  // workspace carve (256B aligned)
  char* ws = (char*)d_ws;
  size_t off = 0;
  auto carve = [&](size_t bytes) -> char* {
    char* p = ws + off;
    off = (off + bytes + 255) & ~(size_t)255;
    return p;
  };
  _Float16* q_short = (_Float16*)carve((size_t)B * SHORT_T * EMB * 2);
  _Float16* i_short = (_Float16*)carve((size_t)B * SHORT_T * EMB * 2);
  _Float16* q_long  = (_Float16*)carve((size_t)B * LONG_T * EMB * 2);
  _Float16* i_long  = (_Float16*)carve((size_t)B * LONG_T * EMB * 2);
  float*    q_self  = (float*)carve((size_t)B * EMB * 4);
  float*    w_eff   = (float*)carve(128 * 4);
  float*    b_eff   = (float*)carve(4);
  (void)ws_size; (void)n_in;

  float* out = (float*)d_out;
  (void)out_size;

  k_att_collapse<<<1, 128, 0, stream>>>(att_W1, att_b1, att_W2, att_b2, w_eff, b_eff);

  size_t smem1 = (size_t)31 * BROWS * EMB * 2;  // 63488 B
  k_pool_qproj<<<nblk, 256, smem1, stream>>>(
      query_ids, query_seg, query_seg_lens, query_click, query_click_lens,
      query_long_his, query_short_his, word_emb, item_emb, Wq, bq,
      q_short, i_short, q_long, i_long, q_self);

  // short: x16(4096)+h16(2048)+gi(12288)+gh(12288)+hbuf(4096)+interest(4096)
  //        +lastt(64)+i16b(2048) = 41024
  size_t smem2 = 48 * 1024;
  k_gru_short<<<nblk, 384, smem2, stream>>>(
      query_ids, short_lens, q_short, i_short,
      s_Wih, s_Whh, s_bih, s_bhh, short_proj, out);

  // long: x16(4096)+h16(2048)+gi(12288)+gh(12288)+hbuf(4096)+e_a(1280)
  //       +i16b(2048)+hist16(40960) = 79104
  size_t smem3 = 84 * 1024;
  k_gru_long<<<nblk, 384, smem3, stream>>>(
      query_ids, long_lens, q_long, i_long,
      l_Wih, l_Whh, l_bih, l_bhh, q_self, w_eff, b_eff, long_proj, out);
}
